// GraphDenoiser_87222195848111
// MI455X (gfx1250) — matrix operand order repaired and checked by reading, not compiled, not run on hardware
//
#include <hip/hip_runtime.h>
#include <hip/hip_bf16.h>
#include <math.h>

typedef __attribute__((ext_vector_type(16))) _Float16 v16h;
typedef __attribute__((ext_vector_type(8)))  _Float16 v8h;
typedef __attribute__((ext_vector_type(8)))  float    v8f;

constexpr int Bz = 4, Nn = 256, Dd = 512, Hh = 8, Ll = 6, TDt = 128, KNk = 16, KEe = 4;
constexpr int DHd = Dd / Hh;           // 64
constexpr int Mrow = Bz * Nn;          // 1024
constexpr float EPSf = 1e-5f;
constexpr size_t MD = (size_t)Mrow * Dd;   // 524288

// ---------------- float workspace layout (float offsets) ----------------
constexpr size_t OFF_H     = 0;                  // B*N*D (residual stream, f32)
constexpr size_t OFF_V     = OFF_H    + MD;      // V f32 (for head transpose)
constexpr size_t OFF_SC    = OFF_V    + MD;      // raw attn logits f32 (4*MD) | pa,pb in pair phase
constexpr size_t OFF_BIAS  = OFF_SC   + 4 * MD;  // attn bias f32 (4*MD)      | etmp in pair phase
constexpr size_t OFF_TRAW  = OFF_BIAS + 4 * MD;
constexpr size_t OFF_TMID  = OFF_TRAW + (size_t)Bz * TDt;
constexpr size_t OFF_TEMB  = OFF_TMID + (size_t)Bz * Dd;
constexpr size_t FLOAT_TOT = OFF_TEMB + (size_t)Bz * Dd;

// ---------------- f16 workspace (half offsets from FLOAT_TOT floats) ----------------
constexpr size_t HW_DxD    = (size_t)Dd * Dd;    // 262144
constexpr size_t HW_LS     = 12 * HW_DxD;        // per-layer weight stride (halves)
constexpr size_t HOFF_WQ   = 0;
constexpr size_t HOFF_WK   = HW_DxD;
constexpr size_t HOFF_WV   = 2 * HW_DxD;
constexpr size_t HOFF_WO   = 3 * HW_DxD;
constexpr size_t HOFF_FF1  = 4 * HW_DxD;         // 4D x D  (transposed ff_w1)
constexpr size_t HOFF_FF2  = 8 * HW_DxD;         // D x 4D  (transposed ff_w2)
constexpr size_t HOFF_NODE = 6 * HW_LS;          // KN x D
constexpr size_t HOFF_PWA  = HOFF_NODE + (size_t)KNk * Dd;
constexpr size_t HOFF_PWB  = HOFF_PWA + HW_DxD;
constexpr size_t HOFF_VT   = HOFF_PWB + HW_DxD;  // B*H*DH*N
constexpr size_t HOFF_QH   = HOFF_VT  + MD;
constexpr size_t HOFF_KH   = HOFF_QH  + MD;
constexpr size_t HOFF_OH   = HOFF_KH  + MD;
constexpr size_t HOFF_HNH  = HOFF_OH  + MD;
constexpr size_t HOFF_HH   = HOFF_HNH + MD;
constexpr size_t HOFF_MIDH = HOFF_HH  + MD;      // B*N*4D
constexpr size_t HOFF_SCH  = HOFF_MIDH + 4 * MD; // f16 softmax probs (B*H*N*N)

__device__ __forceinline__ float silu_f(float x) { return x / (1.f + __expf(-x)); }

// ------------- WMMA fragment loaders (ISA 7.12.2 layouts) -------------
// A: 16x32 (MxK) from row-major f16, lda = row stride; per lane: two contiguous 16B chunks
__device__ __forceinline__ v16h load_ah_frag(const _Float16* __restrict__ A, int row0, int k0, int lda) {
  int lane = threadIdx.x & 31;
  int g = lane >> 4, m = lane & 15;
  const _Float16* base = A + (size_t)(row0 + m) * lda + k0 + g * 8;
  v8h lo = *(const v8h*)(base);         // K = g*8 + e
  v8h hi = *(const v8h*)(base + 16);    // K = 16 + g*8 + e
  v16h a;
#pragma unroll
  for (int e = 0; e < 8; ++e) { a[e] = lo[e]; a[8 + e] = hi[e]; }
  return a;
}

// B: 32x16 (KxN) from f16 N-major (transposed) source: B[k][n] = Xt[col+n][k]
// element e <-> K = g*8 + e (e<8), 16 + g*8 + (e-8) (e>=8) => two 16B loads per lane
__device__ __forceinline__ v16h load_bh_frag(const _Float16* __restrict__ Xt, int col, int k0, int ld) {
  int lane = threadIdx.x & 31;
  int g = lane >> 4, n = lane & 15;
  const _Float16* base = Xt + (size_t)(col + n) * ld + k0 + g * 8;
  v8h lo = *(const v8h*)(base);
  v8h hi = *(const v8h*)(base + 16);
  v16h b;
#pragma unroll
  for (int e = 0; e < 8; ++e) { b[e] = lo[e]; b[8 + e] = hi[e]; }
  return b;
}

// D store, f32 destination
__device__ __forceinline__ void store_d(float* __restrict__ C, v8f acc, int row0, int col0, int ldc) {
  int lane = threadIdx.x & 31;
  int g = lane >> 4, n = lane & 15;
#pragma unroll
  for (int r = 0; r < 8; ++r) C[(size_t)(row0 + r + g * 8) * ldc + col0 + n] = acc[r];
}

// D store, f16 destination
__device__ __forceinline__ void store_dh(_Float16* __restrict__ C, v8f acc, int row0, int col0, int ldc) {
  int lane = threadIdx.x & 31;
  int g = lane >> 4, n = lane & 15;
#pragma unroll
  for (int r = 0; r < 8; ++r) C[(size_t)(row0 + r + g * 8) * ldc + col0 + n] = (_Float16)acc[r];
}

// -------- weight convert + transpose: W (K x Nc, f32) -> WT (Nc x K, f16) --------
__global__ void k_w2hT(const float* __restrict__ W, _Float16* __restrict__ WT, int K, int Nc) {
  size_t idx = (size_t)blockIdx.x * blockDim.x + threadIdx.x;  // K*Nc, coalesced read
  int k = idx / Nc;
  int n = idx - (size_t)k * Nc;
  WT[(size_t)n * K + k] = (_Float16)W[idx];
}

// -------- V head-transpose: v (B,N,D f32) -> vt[(b*H+h)*DH + d][m] f16 --------
__global__ void k_vT(const float* __restrict__ v, _Float16* __restrict__ vt) {
  size_t idx = (size_t)blockIdx.x * blockDim.x + threadIdx.x;  // B*N*D
  int m  = idx % Nn;
  int d  = (idx / Nn) % DHd;
  int bh = idx / ((size_t)Nn * DHd);
  int b  = bh / Hh, hh = bh - b * Hh;
  vt[idx] = (_Float16)v[((size_t)b * Nn + m) * Dd + hh * DHd + d];
}

// ---------------- WMMA GEMM: out = act(A@W + bias) + residual ----------------
// A: Mr x K f16; WT: Nc x K f16 (transposed); each wave: 16 x (16*NW) tile
// Cf (f32) and/or Ch (f16) outputs, either may be null.
template <int NW>
__global__ __launch_bounds__(256) void k_gemm(const _Float16* __restrict__ A,
                                              const _Float16* __restrict__ WT,
                                              const float* __restrict__ bias,
                                              const float* __restrict__ residual,
                                              float* __restrict__ Cf,
                                              _Float16* __restrict__ Ch,
                                              int Mr, int K, int Nc, int act) {
  int wid = blockIdx.x * (blockDim.x >> 5) + (threadIdx.x >> 5);
  int tilesN = Nc / (16 * NW);
  int tM = wid / tilesN, tN = wid - tM * tilesN;
  if (tM * 16 >= Mr) return;                    // wave-uniform; EXEC stays all-ones
  int row0 = tM * 16, col0 = tN * 16 * NW;
  int lane = threadIdx.x & 31;
  const _Float16* arow = A + (size_t)(row0 + (lane & 15)) * K;
  v8f acc[NW];
#pragma unroll
  for (int w = 0; w < NW; ++w) acc[w] = {};
  for (int k0 = 0; k0 < K; k0 += 128) {         // 4 WMMA K-steps / iteration
    __builtin_prefetch(arow + k0 + 128, 0, 3);  // global_prefetch_b8 next A strip
#pragma unroll
    for (int ku = 0; ku < 4; ++ku) {
      int kk = k0 + ku * 32;
      v16h a = load_ah_frag(A, row0, kk, K);
#pragma unroll
      for (int w = 0; w < NW; ++w) {
        v16h b = load_bh_frag(WT, col0 + w * 16, kk, K);
        acc[w] = __builtin_amdgcn_wmma_f32_16x16x32_f16(false, a, false, b, (short)0,
                                                        acc[w], false, false);
      }
    }
  }
  int g = lane >> 4, n = lane & 15;
#pragma unroll
  for (int w = 0; w < NW; ++w) {
    int cb = col0 + w * 16;
#pragma unroll
    for (int r = 0; r < 8; ++r) {
      float vv = acc[w][r];
      if (bias) vv += bias[cb + n];
      if (act == 1) vv = silu_f(vv);
      size_t idx = (size_t)(row0 + r + g * 8) * Nc + cb + n;
      if (residual) vv += residual[idx];
      if (Cf) Cf[idx] = vv;
      if (Ch) Ch[idx] = (_Float16)vv;
    }
  }
}

// ---------------- attention: logits = Q @ K^T per (b,h), f16 operands ----------------
__global__ __launch_bounds__(256) void k_attn_logits(const _Float16* __restrict__ qh,
                                                     const _Float16* __restrict__ kh,
                                                     float* __restrict__ scores) {
  const int TN = Nn / 16;                       // 16
  int wid = blockIdx.x * (blockDim.x >> 5) + (threadIdx.x >> 5);
  int bh = wid / (TN * TN);
  int t  = wid - bh * (TN * TN);
  int ti = t / TN, tj = t - ti * TN;
  int b = bh / Hh, hh = bh - b * Hh;
  const _Float16* Qb = qh + (size_t)b * Nn * Dd + hh * DHd;
  const _Float16* Kb = kh + (size_t)b * Nn * Dd + hh * DHd;
  v8f acc = {};
#pragma unroll
  for (int k0 = 0; k0 < DHd; k0 += 32) {
    v16h a  = load_ah_frag(Qb, ti * 16, k0, Dd);
    v16h bb = load_bh_frag(Kb, tj * 16, k0, Dd);   // B = K^T: contiguous per lane
    acc = __builtin_amdgcn_wmma_f32_16x16x32_f16(false, a, false, bb, (short)0, acc, false, false);
  }
  store_d(scores + (size_t)bh * Nn * Nn, acc, ti * 16, tj * 16, Nn);
}

// ---------------- softmax (scale + bias fused), f32 in -> f16 probs out ----------------
__global__ __launch_bounds__(256) void k_softmax(const float* __restrict__ scores,
                                                 const float* __restrict__ bias4,
                                                 _Float16* __restrict__ probs,
                                                 float scale) {
  int row  = blockIdx.x * (blockDim.x >> 5) + (threadIdx.x >> 5);  // B*H*N rows
  int lane = threadIdx.x & 31;
  const float* p  = scores + (size_t)row * Nn;
  const float* bp = bias4 + (size_t)row * Nn;
  _Float16* ph = probs + (size_t)row * Nn;
  float vals[8];
  float mx = -1e30f;
#pragma unroll
  for (int i = 0; i < 8; ++i) {
    float v = p[lane + i * 32] * scale + bp[lane + i * 32];
    vals[i] = v; mx = fmaxf(mx, v);
  }
#pragma unroll
  for (int o = 16; o; o >>= 1) mx = fmaxf(mx, __shfl_xor(mx, o, 32));
  float s = 0.f;
#pragma unroll
  for (int i = 0; i < 8; ++i) { vals[i] = __expf(vals[i] - mx); s += vals[i]; }
#pragma unroll
  for (int o = 16; o; o >>= 1) s += __shfl_xor(s, o, 32);
  float inv = 1.f / s;
#pragma unroll
  for (int i = 0; i < 8; ++i) ph[lane + i * 32] = (_Float16)(vals[i] * inv);
}

// ---------------- attention: O = P @ V per (b,h); all-f16 operands, f16 out ----------------
__global__ __launch_bounds__(256) void k_attn_av(const _Float16* __restrict__ probs,
                                                 const _Float16* __restrict__ vt,
                                                 _Float16* __restrict__ oh) {
  const int TI = Nn / 16, TJ = DHd / 16;        // 16, 4
  int wid = blockIdx.x * (blockDim.x >> 5) + (threadIdx.x >> 5);
  int bh = wid / (TI * TJ);
  int t  = wid - bh * (TI * TJ);
  int ti = t / TJ, tj = t - ti * TJ;
  int b = bh / Hh, hh = bh - b * Hh;
  const _Float16* P  = probs + (size_t)bh * Nn * Nn;
  const _Float16* Vt = vt + (size_t)bh * DHd * Nn;   // [d][m] layout
  v8f acc = {};
#pragma unroll
  for (int k0 = 0; k0 < Nn; k0 += 32) {         // 8 back-to-back WMMAs
    v16h a  = load_ah_frag(P, ti * 16, k0, Nn);
    v16h bb = load_bh_frag(Vt, tj * 16, k0, Nn);
    acc = __builtin_amdgcn_wmma_f32_16x16x32_f16(false, a, false, bb, (short)0, acc, false, false);
  }
  store_dh(oh + (size_t)b * Nn * Dd + hh * DHd, acc, ti * 16, tj * 16, Dd);
}

// ---------------- layernorm: wave per row of D=512, f32 in -> f16 out ----------------
__global__ __launch_bounds__(256) void k_ln(const float* __restrict__ x,
                                            const float* __restrict__ g,
                                            const float* __restrict__ bta,
                                            _Float16* __restrict__ out) {
  int row  = blockIdx.x * (blockDim.x >> 5) + (threadIdx.x >> 5);
  int lane = threadIdx.x & 31;
  const float* xr = x + (size_t)row * Dd;
  float loc[16]; float s = 0.f, s2 = 0.f;
#pragma unroll
  for (int i = 0; i < 16; ++i) {
    float v = xr[lane + i * 32]; loc[i] = v; s += v; s2 += v * v;
  }
#pragma unroll
  for (int o = 16; o; o >>= 1) { s += __shfl_xor(s, o, 32); s2 += __shfl_xor(s2, o, 32); }
  float mean = s * (1.f / Dd);
  float var  = s2 * (1.f / Dd) - mean * mean;
  float r = rsqrtf(var + EPSf);
  _Float16* orow = out + (size_t)row * Dd;
#pragma unroll
  for (int i = 0; i < 16; ++i) {
    int d = lane + i * 32;
    orow[d] = (_Float16)((loc[i] - mean) * r * g[d] + bta[d]);
  }
}

// ---------------- tiny time-embedding MLP (M=B=4) ----------------
__global__ void k_mlp(const float* __restrict__ in, const float* __restrict__ W,
                      const float* __restrict__ bias, float* __restrict__ out,
                      int Kdim, int act) {
  int idx = blockIdx.x * blockDim.x + threadIdx.x;   // B*D
  int b = idx / Dd, dout = idx - b * Dd;
  float s = bias[dout];
  for (int kk = 0; kk < Kdim; ++kk) s += in[b * Kdim + kk] * W[(size_t)kk * Dd + dout];
  if (act) s = silu_f(s);
  out[idx] = s;
}

__global__ void k_time_raw(const int* __restrict__ t, float* __restrict__ raw) {
  int idx = blockIdx.x * blockDim.x + threadIdx.x;   // B*TD
  int b = idx / TDt, j = idx - b * TDt;
  const int half = TDt / 2;
  int jj = (j < half) ? j : j - half;
  float freq = __expf(-9.210340371976184f * (float)jj / (float)half);
  float ang = (float)t[b] * freq;
  raw[idx] = (j < half) ? __sinf(ang) : __cosf(ang);
}

__global__ void k_init_h(const int* __restrict__ xn, const float* __restrict__ emb,
                         const float* __restrict__ temb, const float* __restrict__ cemb,
                         float* __restrict__ h) {
  size_t idx = (size_t)blockIdx.x * blockDim.x + threadIdx.x;   // B*N*D
  int d = idx % Dd;
  int bn = idx / Dd;
  int b = bn / Nn;
  int id = xn[bn];
  h[idx] = emb[(size_t)id * Dd + d] + temb[b * Dd + d] + cemb[b * Dd + d];
}

__global__ void k_bias_gather(const int* __restrict__ xe, const float* __restrict__ tab,
                              float* __restrict__ bias4) {
  size_t idx = (size_t)blockIdx.x * blockDim.x + threadIdx.x;   // B*H*N*N
  int j  = idx % Nn;
  int i  = (idx / Nn) % Nn;
  int hh = (idx / ((size_t)Nn * Nn)) % Hh;
  int b  = idx / ((size_t)Hh * Nn * Nn);
  int e  = xe[((size_t)b * Nn + i) * Nn + j];
  bias4[idx] = tab[e * Hh + hh];
}

// ---------------- pair head: etmp[b,i,j,:] = silu(pa_i+pb_j+b1) @ w2 + b2 ----------------
__global__ __launch_bounds__(256) void k_pair(const float* __restrict__ pa,
                                              const float* __restrict__ pb,
                                              const float* __restrict__ b1,
                                              const float* __restrict__ w2,
                                              const float* __restrict__ b2,
                                              float* __restrict__ etmp) {
  int wid  = blockIdx.x * (blockDim.x >> 5) + (threadIdx.x >> 5);  // b*N*N + i*N + j
  int lane = threadIdx.x & 31;
  int b = wid / (Nn * Nn);
  int rem = wid - b * (Nn * Nn);
  int i = rem / Nn, j = rem - i * Nn;
  const float* par = pa + ((size_t)b * Nn + i) * Dd;
  const float* pbr = pb + ((size_t)b * Nn + j) * Dd;
  float a0 = 0.f, a1 = 0.f, a2 = 0.f, a3 = 0.f;
#pragma unroll 4
  for (int it = 0; it < 16; ++it) {
    int d = lane + it * 32;
    float s = silu_f(par[d] + pbr[d] + b1[d]);
    float4 w = *(const float4*)(w2 + (size_t)d * 4);
    a0 += s * w.x; a1 += s * w.y; a2 += s * w.z; a3 += s * w.w;
  }
#pragma unroll
  for (int o = 16; o; o >>= 1) {
    a0 += __shfl_xor(a0, o, 32); a1 += __shfl_xor(a1, o, 32);
    a2 += __shfl_xor(a2, o, 32); a3 += __shfl_xor(a3, o, 32);
  }
  if (lane == 0) {
    float4 r; r.x = a0 + b2[0]; r.y = a1 + b2[1]; r.z = a2 + b2[2]; r.w = a3 + b2[3];
    *(float4*)(etmp + (size_t)wid * 4) = r;
  }
}

__global__ void k_sym(const float* __restrict__ etmp, float* __restrict__ out) {
  size_t idx = (size_t)blockIdx.x * blockDim.x + threadIdx.x;   // B*N*N*KE
  int e = idx & 3;
  size_t pij = idx >> 2;
  int b = pij / (Nn * Nn);
  int rem = pij - (size_t)b * Nn * Nn;
  int i = rem / Nn, j = rem - i * Nn;
  float v;
  if (i == j) {
    v = (e == 0) ? 1e9f : -1e9f;
  } else {
    float x = etmp[idx];
    float y = etmp[(((size_t)b * Nn + j) * Nn + i) * 4 + e];
    v = 0.5f * (x + y);
  }
  out[idx] = v;
}

// =======================================================================
extern "C" void kernel_launch(void* const* d_in, const int* in_sizes, int n_in,
                              void* d_out, int out_size, void* d_ws, size_t ws_size,
                              hipStream_t stream) {
  const int*   x_nodes = (const int*)d_in[0];
  const int*   x_edges = (const int*)d_in[1];
  const int*   t       = (const int*)d_in[2];
  const float* c_emb   = (const float*)d_in[3];
  const float* nemb    = (const float*)d_in[4];
  const float* tw1     = (const float*)d_in[5];
  const float* tb1     = (const float*)d_in[6];
  const float* tw2     = (const float*)d_in[7];
  const float* tb2     = (const float*)d_in[8];
  const float* ebias   = (const float*)d_in[9];
  const float* ln1g    = (const float*)d_in[10];
  const float* ln1b    = (const float*)d_in[11];
  const float* Wq      = (const float*)d_in[12];
  const float* Wk      = (const float*)d_in[13];
  const float* Wv      = (const float*)d_in[14];
  const float* Wo      = (const float*)d_in[15];
  const float* ln2g    = (const float*)d_in[16];
  const float* ln2b    = (const float*)d_in[17];
  const float* ffw1    = (const float*)d_in[18];
  const float* ffb1    = (const float*)d_in[19];
  const float* ffw2    = (const float*)d_in[20];
  const float* ffb2    = (const float*)d_in[21];
  const float* node_w  = (const float*)d_in[22];
  const float* node_b  = (const float*)d_in[23];
  const float* pw1     = (const float*)d_in[24];
  const float* pb1     = (const float*)d_in[25];
  const float* pw2     = (const float*)d_in[26];
  const float* pb2     = (const float*)d_in[27];

  float* ws    = (float*)d_ws;
  float* h     = ws + OFF_H;
  float* v     = ws + OFF_V;
  float* sc    = ws + OFF_SC;        // attention phase (raw logits)
  float* bias4 = ws + OFF_BIAS;
  float* traw  = ws + OFF_TRAW;
  float* tmid  = ws + OFF_TMID;
  float* temb  = ws + OFF_TEMB;
  float* pa    = ws + OFF_SC;        // pair phase aliases (sc/bias4 dead by then)
  float* pb    = ws + OFF_SC + MD;
  float* etmp  = ws + OFF_BIAS;

  _Float16* hw    = (_Float16*)(ws + FLOAT_TOT);
  _Float16* vt    = hw + HOFF_VT;
  _Float16* qh    = hw + HOFF_QH;
  _Float16* kh    = hw + HOFF_KH;
  _Float16* oh    = hw + HOFF_OH;
  _Float16* hnh   = hw + HOFF_HNH;
  _Float16* hh16  = hw + HOFF_HH;
  _Float16* midh  = hw + HOFF_MIDH;
  _Float16* sch   = hw + HOFF_SCH;

  float* node_logits = (float*)d_out;                 // B*N*KN
  float* edge_out    = (float*)d_out + Bz * Nn * KNk; // B*N*N*KE

  const int TB = 256;                                 // 8 waves / block
  auto gblk = [](int Mr, int Nc, int NW) { return ((Mr / 16) * (Nc / (16 * NW))) / 8; };

  // --- one-pass weight convert + transpose to f16 (L2-resident afterwards) ---
  for (int l = 0; l < Ll; ++l) {
    const size_t wD = (size_t)l * Dd * Dd;
    _Float16* hl = hw + (size_t)l * HW_LS;
    k_w2hT<<<(Dd * Dd) / TB, TB, 0, stream>>>(Wq + wD, hl + HOFF_WQ, Dd, Dd);
    k_w2hT<<<(Dd * Dd) / TB, TB, 0, stream>>>(Wk + wD, hl + HOFF_WK, Dd, Dd);
    k_w2hT<<<(Dd * Dd) / TB, TB, 0, stream>>>(Wv + wD, hl + HOFF_WV, Dd, Dd);
    k_w2hT<<<(Dd * Dd) / TB, TB, 0, stream>>>(Wo + wD, hl + HOFF_WO, Dd, Dd);
    k_w2hT<<<(Dd * 4 * Dd) / TB, TB, 0, stream>>>(ffw1 + (size_t)l * Dd * 4 * Dd, hl + HOFF_FF1, Dd, 4 * Dd);
    k_w2hT<<<(4 * Dd * Dd) / TB, TB, 0, stream>>>(ffw2 + (size_t)l * 4 * Dd * Dd, hl + HOFF_FF2, 4 * Dd, Dd);
  }
  k_w2hT<<<(Dd * KNk) / TB, TB, 0, stream>>>(node_w, hw + HOFF_NODE, Dd, KNk);
  k_w2hT<<<(Dd * Dd) / TB, TB, 0, stream>>>(pw1, hw + HOFF_PWA, Dd, Dd);
  k_w2hT<<<(Dd * Dd) / TB, TB, 0, stream>>>(pw1 + (size_t)Dd * Dd, hw + HOFF_PWB, Dd, Dd);

  // --- time embedding ---
  k_time_raw<<<(Bz * TDt) / TB, TB, 0, stream>>>(t, traw);
  k_mlp<<<(Bz * Dd) / TB, TB, 0, stream>>>(traw, tw1, tb1, tmid, TDt, 1);
  k_mlp<<<(Bz * Dd) / TB, TB, 0, stream>>>(tmid, tw2, tb2, temb, Dd, 0);

  // --- h init + attention bias gather ---
  k_init_h<<<(Mrow * Dd) / TB, TB, 0, stream>>>(x_nodes, nemb, temb, c_emb, h);
  k_bias_gather<<<(Bz * Hh * Nn * Nn) / TB, TB, 0, stream>>>(x_edges, ebias, bias4);

  const float scale = 0.125f;                         // DH^-0.5
  for (int l = 0; l < Ll; ++l) {
    _Float16* hl = hw + (size_t)l * HW_LS;
    // attention block
    k_ln<<<Mrow / 8, TB, 0, stream>>>(h, ln1g + l * Dd, ln1b + l * Dd, hnh);
    k_gemm<2><<<gblk(Mrow, Dd, 2), TB, 0, stream>>>(hnh, hl + HOFF_WQ, nullptr, nullptr,
                                                    nullptr, qh, Mrow, Dd, Dd, 0);
    k_gemm<2><<<gblk(Mrow, Dd, 2), TB, 0, stream>>>(hnh, hl + HOFF_WK, nullptr, nullptr,
                                                    nullptr, kh, Mrow, Dd, Dd, 0);
    k_gemm<2><<<gblk(Mrow, Dd, 2), TB, 0, stream>>>(hnh, hl + HOFF_WV, nullptr, nullptr,
                                                    v, nullptr, Mrow, Dd, Dd, 0);
    k_vT<<<(Mrow * Dd) / TB, TB, 0, stream>>>(v, vt);
    k_attn_logits<<<(Bz * Hh * (Nn / 16) * (Nn / 16)) / 8, TB, 0, stream>>>(qh, kh, sc);
    k_softmax<<<(Bz * Hh * Nn) / 8, TB, 0, stream>>>(sc, bias4, sch, scale);
    k_attn_av<<<(Bz * Hh * (Nn / 16) * (DHd / 16)) / 8, TB, 0, stream>>>(sch, vt, oh);
    k_gemm<2><<<gblk(Mrow, Dd, 2), TB, 0, stream>>>(oh, hl + HOFF_WO, nullptr, h,
                                                    h, nullptr, Mrow, Dd, Dd, 0);
    // FFN block
    k_ln<<<Mrow / 8, TB, 0, stream>>>(h, ln2g + l * Dd, ln2b + l * Dd, hnh);
    k_gemm<2><<<gblk(Mrow, 4 * Dd, 2), TB, 0, stream>>>(hnh, hl + HOFF_FF1,
                                                        ffb1 + (size_t)l * 4 * Dd, nullptr,
                                                        nullptr, midh, Mrow, Dd, 4 * Dd, 1);
    k_gemm<2><<<gblk(Mrow, Dd, 2), TB, 0, stream>>>(midh, hl + HOFF_FF2,
                                                    ffb2 + (size_t)l * Dd, h,
                                                    h, hh16, Mrow, 4 * Dd, Dd, 0);
  }

  // --- node logits (Nc = 16 -> single-tile variant) ---
  k_gemm<1><<<gblk(Mrow, KNk, 1), TB, 0, stream>>>(hh16, hw + HOFF_NODE, node_b, nullptr,
                                                   node_logits, nullptr, Mrow, Dd, KNk, 0);

  // --- pair head ---
  k_gemm<2><<<gblk(Mrow, Dd, 2), TB, 0, stream>>>(hh16, hw + HOFF_PWA, nullptr, nullptr,
                                                  pa, nullptr, Mrow, Dd, Dd, 0);
  k_gemm<2><<<gblk(Mrow, Dd, 2), TB, 0, stream>>>(hh16, hw + HOFF_PWB, nullptr, nullptr,
                                                  pb, nullptr, Mrow, Dd, Dd, 0);
  k_pair<<<(Bz * Nn * Nn) / 8, TB, 0, stream>>>(pa, pb, pb1, pw2, pb2, etmp);
  k_sym<<<(Bz * Nn * Nn * KEe) / TB, TB, 0, stream>>>(etmp, edge_out);
}
